// GPSModel_64484638982372
// MI455X (gfx1250) — matrix operand order repaired and checked by reading, not compiled, not run on hardware
//
#include <hip/hip_runtime.h>
#include <math.h>

// ---------------------------------------------------------------------------
// GPS (GraphGPS) forward for MI455X / gfx1250.
// GEMMs: bf16 WMMA 16x16x32 -> f32 accum, with B pre-packed into fragment
// layout (2x b128 loads per fragment, linear addressing). Graph ops: atomics.
// ---------------------------------------------------------------------------

#define NODES  50000
#define EDGES  600000
#define FD     128
#define HEADS  4
#define DHD    32
#define LAYERS 3
#define CLS    10
#define EPSV   1e-5f

typedef __attribute__((ext_vector_type(16))) __bf16 v16bf;
typedef __attribute__((ext_vector_type(8)))  __bf16 v8bf;
typedef __attribute__((ext_vector_type(8)))  float  v8f;

// ---------------- helpers ----------------

static __device__ __forceinline__ __bf16 f2bf(float f) {
  unsigned u = __float_as_uint(f);
  u += 0x7FFFu + ((u >> 16) & 1u);   // round-to-nearest-even truncation
  unsigned short s = (unsigned short)(u >> 16);
  __bf16 r;
  __builtin_memcpy(&r, &s, 2);
  return r;
}

static __device__ __forceinline__ unsigned f2ord(float f) {
  unsigned u = __float_as_uint(f);
  return (u & 0x80000000u) ? ~u : (u | 0x80000000u);
}
static __device__ __forceinline__ float ord2f(unsigned e) {
  unsigned u = (e & 0x80000000u) ? (e ^ 0x80000000u) : ~e;
  return __uint_as_float(u);
}

static __device__ __forceinline__ float wred(float v) {
#pragma unroll
  for (int o = 16; o; o >>= 1) v += __shfl_xor(v, o, 32);
  return v;
}

static __device__ __forceinline__ v16bf cat16(v8bf lo, v8bf hi) {
  return __builtin_shufflevector(lo, hi, 0, 1, 2, 3, 4, 5, 6, 7,
                                 8, 9, 10, 11, 12, 13, 14, 15);
}

// ---------------- tiny utility kernels ----------------

__global__ void k_zero(float* p, int n) {
  int i = blockIdx.x * blockDim.x + threadIdx.x;
  if (i < n) p[i] = 0.f;
}
__global__ void k_fill_u32(unsigned* p, unsigned v, int n) {
  int i = blockIdx.x * blockDim.x + threadIdx.x;
  if (i < n) p[i] = v;
}
__global__ void k_cvt_bf16(const float* __restrict__ s, __bf16* __restrict__ d, int n) {
  int i = blockIdx.x * blockDim.x + threadIdx.x;
  if (i < n) d[i] = f2bf(s[i]);
}

// Pack a row-major f32 weight W[K,Ncol] into bf16 WMMA B-fragment layout:
// P[((kt*NT + nt)*32 + lane)*16 + e]  where h=lane>>4, K=kt*32+8h+(e&7)+(e>>3)*16,
//                                          N=nt*16+(lane&15)
__global__ void k_pack_b(const float* __restrict__ W, __bf16* __restrict__ P,
                         int K, int Ncol) {
  int i = blockIdx.x * blockDim.x + threadIdx.x;
  if (i >= K * Ncol) return;
  int e    = i & 15;
  int lane = (i >> 4) & 31;
  int tile = i >> 9;
  int NT   = Ncol >> 4;
  int kt   = tile / NT, nt = tile % NT;
  int k = kt * 32 + ((lane >> 4) << 3) + (e & 7) + ((e >> 3) << 4);
  int n = nt * 16 + (lane & 15);
  P[i] = f2bf(W[(size_t)k * Ncol + n]);
}

// ---------------- WMMA GEMM:  C[M,Ncol] = A[M,K](bf16) x Bp(packed bf16) ----
// One wave -> one 16-row tile x 128 columns (8 col-tiles), K stepped by 32.
// A fragment per lane = two contiguous 16B runs (ISA 16-bit A 16x32 layout).

__global__ void k_wmma_gemm(const __bf16* __restrict__ A, const __bf16* __restrict__ Bp,
                            const float* __restrict__ bias, float* __restrict__ C,
                            __bf16* __restrict__ Cbf, int M, int K, int Ncol, int relu) {
  const int lane = threadIdx.x & 31;
  const int wave = threadIdx.x >> 5;
  const int tile = blockIdx.x * 4 + wave;
  if (tile * 16 >= M) return;                 // wave-uniform guard
  const int row0 = tile * 16;
  const int NT   = Ncol >> 4;                 // total col tiles in packed B
  const int nt0  = blockIdx.y * 8;            // this block's first col tile

  const int m  = lane & 15;
  const int kh = (lane >> 4) << 3;            // A per-lane-half K offset (0|8)

  v8f acc[8];
#pragma unroll
  for (int t = 0; t < 8; ++t)
#pragma unroll
    for (int r = 0; r < 8; ++r) acc[t][r] = 0.f;

  const __bf16* arow = A + (size_t)(row0 + m) * K + kh;
  const int ktn = K >> 5;
  for (int kt = 0; kt < ktn; ++kt) {
    const __bf16* ap = arow + (kt << 5);
    v16bf a = cat16(*(const v8bf*)ap, *(const v8bf*)(ap + 16));
    const __bf16* bbase = Bp + (((size_t)kt * NT + nt0) * 32 + lane) * 16;
#pragma unroll
    for (int nt = 0; nt < 8; ++nt) {
      const __bf16* bp = bbase + (size_t)nt * 32 * 16;
      v16bf b = cat16(*(const v8bf*)bp, *(const v8bf*)(bp + 8));
      acc[nt] = __builtin_amdgcn_wmma_f32_16x16x32_bf16(
          false, a, false, b, (short)0, acc[nt], false, false);
    }
  }

  const int mh = (lane >> 4) << 3;
  const int bn = lane & 15;
#pragma unroll
  for (int nt = 0; nt < 8; ++nt) {
    const int col = (nt0 + nt) * 16 + bn;
    const float bv = bias ? bias[col] : 0.f;
#pragma unroll
    for (int r = 0; r < 8; ++r) {
      const int row = row0 + mh + r;
      float v = acc[nt][r] + bv;
      if (relu) v = fmaxf(v, 0.f);
      C[(size_t)row * Ncol + col] = v;
      if (Cbf) Cbf[(size_t)row * Ncol + col] = f2bf(v);
    }
  }
}

// ---------------- BatchNorm (training-mode, over node axis) ----------------

__global__ void k_colstats(const float* __restrict__ X, int n,
                           float* __restrict__ sum, float* __restrict__ sumsq) {
  int c = threadIdx.x;                         // blockDim.x == FD
  float s = 0.f, q = 0.f;
  for (int r = blockIdx.x; r < n; r += gridDim.x) {
    float v = X[(size_t)r * FD + c];
    s += v; q += v * v;
  }
  atomicAdd(&sum[c], s);
  atomicAdd(&sumsq[c], q);
}

__global__ void k_bn_relu(const float* __restrict__ X, const float* __restrict__ sum,
                          const float* __restrict__ sumsq, const float* __restrict__ sc,
                          const float* __restrict__ bi, float* __restrict__ Y,
                          __bf16* __restrict__ Ybf, int n) {
  int i = blockIdx.x * blockDim.x + threadIdx.x;
  if (i >= n * FD) return;
  int c = i & (FD - 1);
  float invn = 1.f / (float)n;
  float m = sum[c] * invn;
  float v = sumsq[c] * invn - m * m;
  float w = rsqrtf(fmaxf(v, 0.f) + EPSV);
  float val = (X[i] - m) * w * sc[c] + bi[c];
  val = fmaxf(val, 0.f);
  Y[i] = val;
  if (Ybf) Ybf[i] = f2bf(val);
}

// ---------------- GCN degree / norm / scatter ----------------

__global__ void k_deg(const int* __restrict__ dst, float* __restrict__ deg) {
  int e = blockIdx.x * blockDim.x + threadIdx.x;
  if (e < EDGES) atomicAdd(&deg[dst[e]], 1.f);
}
__global__ void k_dis(const float* __restrict__ deg, float* __restrict__ dis) {
  int i = blockIdx.x * blockDim.x + threadIdx.x;
  if (i >= NODES) return;
  float d = deg[i];
  dis[i] = (d > 0.f) ? rsqrtf(fmaxf(d, 1.f)) : 0.f;
}
__global__ void k_gcn_scatter(const float* __restrict__ hw, const int* __restrict__ src,
                              const int* __restrict__ dst, const float* __restrict__ dis,
                              float* __restrict__ local) {
  int gid = blockIdx.x * blockDim.x + threadIdx.x;
  if (gid >= EDGES * 32) return;
  int e = gid >> 5, c4 = (gid & 31) << 2;
  int s = src[e], d = dst[e];
  float nrm = dis[s] * dis[d];
  const float4 v = *(const float4*)&hw[(size_t)s * FD + c4];
  float* o = &local[(size_t)d * FD + c4];
  atomicAdd(o + 0, v.x * nrm);
  atomicAdd(o + 1, v.y * nrm);
  atomicAdd(o + 2, v.z * nrm);
  atomicAdd(o + 3, v.w * nrm);
}

// ---------------- TransformerConv attention ----------------

__global__ void k_attn_score(const float* __restrict__ q, const float* __restrict__ k,
                             const int* __restrict__ src, const int* __restrict__ dst,
                             float* __restrict__ score, unsigned* __restrict__ smax) {
  int gid = blockIdx.x * blockDim.x + threadIdx.x;
  if (gid >= EDGES * HEADS) return;
  int e = gid >> 2, h = gid & 3;
  int s = src[e], d = dst[e];
  const float4* qp = (const float4*)&q[(size_t)d * FD + h * DHD];
  const float4* kp = (const float4*)&k[(size_t)s * FD + h * DHD];
  float acc = 0.f;
#pragma unroll
  for (int i = 0; i < 8; ++i) {
    float4 a = qp[i], b = kp[i];
    acc += a.x * b.x + a.y * b.y + a.z * b.z + a.w * b.w;
  }
  acc *= 0.1767766952966369f;                 // 1/sqrt(32)
  score[gid] = acc;
  atomicMax(&smax[d * HEADS + h], f2ord(acc));
}

__global__ void k_attn_exp(float* __restrict__ ex, const unsigned* __restrict__ smax,
                           const int* __restrict__ dst, float* __restrict__ denom) {
  int gid = blockIdx.x * blockDim.x + threadIdx.x;
  if (gid >= EDGES * HEADS) return;
  int e = gid >> 2, h = gid & 3;
  int d = dst[e];
  float mx = ord2f(smax[d * HEADS + h]);
  float ev = expf(ex[gid] - mx);
  ex[gid] = ev;
  atomicAdd(&denom[d * HEADS + h], ev);
}

__global__ void k_attn_agg(const float* __restrict__ ex, const float* __restrict__ denom,
                           const float* __restrict__ v, const int* __restrict__ src,
                           const int* __restrict__ dst, float* __restrict__ agg) {
  int gid = blockIdx.x * blockDim.x + threadIdx.x;
  if (gid >= EDGES * 32) return;
  int e = gid >> 5, idx = gid & 31;
  int h = idx >> 3, c4 = (idx & 7) << 2;
  int s = src[e], d = dst[e];
  float a = ex[e * HEADS + h] / denom[d * HEADS + h];
  const float4 vv = *(const float4*)&v[(size_t)s * FD + h * DHD + c4];
  float* o = &agg[(size_t)d * FD + h * DHD + c4];
  atomicAdd(o + 0, vv.x * a);
  atomicAdd(o + 1, vv.y * a);
  atomicAdd(o + 2, vv.z * a);
  atomicAdd(o + 3, vv.w * a);
}

// beta gate: beta = sigmoid([agg, xr, agg-xr] @ bw); glob -> agg (in place)
__global__ void k_beta_glob(const float* __restrict__ xr, float* __restrict__ agg,
                            const float* __restrict__ bw) {
  int lane = threadIdx.x & 31;
  int nid = blockIdx.x * 4 + (threadIdx.x >> 5);
  if (nid >= NODES) return;
  int c = lane * 4;
  float p = 0.f;
#pragma unroll
  for (int j = 0; j < 4; ++j) {
    size_t o = (size_t)nid * FD + c + j;
    float a = agg[o], x = xr[o];
    p += a * bw[c + j] + x * bw[FD + c + j] + (a - x) * bw[2 * FD + c + j];
  }
  p = wred(p);
  float beta = 1.f / (1.f + expf(-p));
#pragma unroll
  for (int j = 0; j < 4; ++j) {
    size_t o = (size_t)nid * FD + c + j;
    agg[o] = beta * xr[o] + (1.f - beta) * agg[o];
  }
}

// y = LN(2*(lw*(local+gcn_b) + gw*glob)) ; writes f32 + bf16
__global__ void k_combine_ln1(const float* __restrict__ local, const float* __restrict__ glob,
                              const float* __restrict__ gcnb, const float* __restrict__ lwp,
                              const float* __restrict__ gwp, int l,
                              const float* __restrict__ s, const float* __restrict__ b,
                              float* __restrict__ y, __bf16* __restrict__ ybf) {
  int lane = threadIdx.x & 31;
  int nid = blockIdx.x * 4 + (threadIdx.x >> 5);
  if (nid >= NODES) return;
  float lw = lwp[l], gw = gwp[l];
  int c = lane * 4;
  float t[4];
#pragma unroll
  for (int j = 0; j < 4; ++j) {
    size_t o = (size_t)nid * FD + c + j;
    t[j] = 2.f * (lw * (local[o] + gcnb[c + j]) + gw * glob[o]);
  }
  float sm = wred(t[0] + t[1] + t[2] + t[3]);
  float mean = sm * (1.f / FD);
  float vq = 0.f;
#pragma unroll
  for (int j = 0; j < 4; ++j) { float d = t[j] - mean; vq += d * d; }
  vq = wred(vq);
  float inv = rsqrtf(vq * (1.f / FD) + EPSV);
#pragma unroll
  for (int j = 0; j < 4; ++j) {
    size_t o = (size_t)nid * FD + c + j;
    float val = (t[j] - mean) * inv * s[c + j] + b[c + j];
    y[o] = val;
    ybf[o] = f2bf(val);
  }
}

// f2 = LN(f2 + y)   (in place on f2)
__global__ void k_add_ln2(float* __restrict__ f2, const float* __restrict__ y,
                          const float* __restrict__ s, const float* __restrict__ b) {
  int lane = threadIdx.x & 31;
  int nid = blockIdx.x * 4 + (threadIdx.x >> 5);
  if (nid >= NODES) return;
  int c = lane * 4;
  float t[4];
#pragma unroll
  for (int j = 0; j < 4; ++j) {
    size_t o = (size_t)nid * FD + c + j;
    t[j] = f2[o] + y[o];
  }
  float sm = wred(t[0] + t[1] + t[2] + t[3]);
  float mean = sm * (1.f / FD);
  float vq = 0.f;
#pragma unroll
  for (int j = 0; j < 4; ++j) { float d = t[j] - mean; vq += d * d; }
  vq = wred(vq);
  float inv = rsqrtf(vq * (1.f / FD) + EPSV);
#pragma unroll
  for (int j = 0; j < 4; ++j) {
    size_t o = (size_t)nid * FD + c + j;
    f2[o] = (t[j] - mean) * inv * s[c + j] + b[c + j];
  }
}

// logits = h @ out_W + out_b ; log_softmax. One wave per node.
__global__ void k_out(const float* __restrict__ h, const float* __restrict__ W,
                      const float* __restrict__ bias, float* __restrict__ out) {
  int lane = threadIdx.x & 31;
  int nid = blockIdx.x * 4 + (threadIdx.x >> 5);
  if (nid >= NODES) return;
  float p[CLS];
#pragma unroll
  for (int cc = 0; cc < CLS; ++cc) p[cc] = 0.f;
  int c = lane * 4;
#pragma unroll
  for (int j = 0; j < 4; ++j) {
    float hv = h[(size_t)nid * FD + c + j];
    const float* wr = &W[(size_t)(c + j) * CLS];
#pragma unroll
    for (int cc = 0; cc < CLS; ++cc) p[cc] += hv * wr[cc];
  }
#pragma unroll
  for (int o = 16; o; o >>= 1)
#pragma unroll
    for (int cc = 0; cc < CLS; ++cc) p[cc] += __shfl_xor(p[cc], o, 32);
  float mx = -1e30f;
#pragma unroll
  for (int cc = 0; cc < CLS; ++cc) { p[cc] += bias[cc]; mx = fmaxf(mx, p[cc]); }
  float se = 0.f;
#pragma unroll
  for (int cc = 0; cc < CLS; ++cc) se += expf(p[cc] - mx);
  float lse = mx + logf(se);
  if (lane == 0) {
#pragma unroll
    for (int cc = 0; cc < CLS; ++cc) out[(size_t)nid * CLS + cc] = p[cc] - lse;
  }
}

// ---------------------------------------------------------------------------

extern "C" void kernel_launch(void* const* d_in, const int* in_sizes, int n_in,
                              void* d_out, int out_size, void* d_ws, size_t ws_size,
                              hipStream_t stream) {
  const float* x        = (const float*)d_in[0];
  const int*   ei       = (const int*)d_in[1];
  const float* proj_W   = (const float*)d_in[2];
  const float* proj_b   = (const float*)d_in[3];
  const float* bn_in_s  = (const float*)d_in[4];
  const float* bn_in_b  = (const float*)d_in[5];
  const float* gcn_W    = (const float*)d_in[6];
  const float* gcn_b    = (const float*)d_in[7];
  const float* q_W      = (const float*)d_in[8];
  const float* q_b      = (const float*)d_in[9];
  const float* k_W      = (const float*)d_in[10];
  const float* k_b      = (const float*)d_in[11];
  const float* v_W      = (const float*)d_in[12];
  const float* v_b      = (const float*)d_in[13];
  const float* skip_W   = (const float*)d_in[14];
  const float* skip_b   = (const float*)d_in[15];
  const float* beta_W   = (const float*)d_in[16];
  const float* ln1_s    = (const float*)d_in[17];
  const float* ln1_b    = (const float*)d_in[18];
  const float* ln2_s    = (const float*)d_in[19];
  const float* ln2_b    = (const float*)d_in[20];
  const float* rel_W    = (const float*)d_in[21];
  const float* root_W   = (const float*)d_in[22];
  const float* lw       = (const float*)d_in[23];
  const float* gw       = (const float*)d_in[24];
  const float* bn_s     = (const float*)d_in[25];
  const float* bn_b     = (const float*)d_in[26];
  const float* out_W    = (const float*)d_in[27];
  const float* out_b    = (const float*)d_in[28];
  float* out = (float*)d_out;

  const int* src = ei;
  const int* dst = ei + EDGES;

  // ---- workspace carve (~340 MB) ----
  char* wp = (char*)d_ws;
  auto carve = [&](size_t bytes) -> void* {
    void* r = (void*)wp;
    wp += (bytes + 255) & ~(size_t)255;
    return r;
  };
  const size_t ND = (size_t)NODES * FD;
  float*  h      = (float*)carve(ND * 4);
  float*  qb_    = (float*)carve(ND * 4);
  float*  kb_    = (float*)carve(ND * 4);
  float*  vb_    = (float*)carve(ND * 4);
  float*  xr     = (float*)carve(ND * 4);
  float*  local  = (float*)carve(ND * 4);
  float*  y      = (float*)carve(ND * 4);
  float*  f2     = (float*)carve(ND * 4);
  float*  f1     = (float*)carve(ND * 2 * 4);     // FFN hidden (N x 256)
  float*  hw     = f1;                            // alias: dead before FFN
  float*  agg    = f1 + ND;                       // alias: dead before FFN
  __bf16* x_bf   = (__bf16*)carve(ND * 2);
  __bf16* h_bf   = (__bf16*)carve(ND * 2);
  __bf16* y_bf   = (__bf16*)carve(ND * 2);
  __bf16* f1_bf  = (__bf16*)carve(ND * 2 * 2);
  float*  ex     = (float*)carve((size_t)EDGES * HEADS * 4);
  unsigned* smax = (unsigned*)carve((size_t)NODES * HEADS * 4);
  float*  denom  = (float*)carve((size_t)NODES * HEADS * 4);
  float*  deg    = (float*)carve((size_t)NODES * 4);
  float*  dis    = (float*)carve((size_t)NODES * 4);
  float*  stats  = (float*)carve(2 * FD * 4);     // colsum | colsumsq
  __bf16* wproj  = (__bf16*)carve((size_t)FD * FD * 2);
  __bf16* wgcn   = (__bf16*)carve((size_t)LAYERS * FD * FD * 2);
  __bf16* wq     = (__bf16*)carve((size_t)LAYERS * FD * FD * 2);
  __bf16* wk     = (__bf16*)carve((size_t)LAYERS * FD * FD * 2);
  __bf16* wv     = (__bf16*)carve((size_t)LAYERS * FD * FD * 2);
  __bf16* wskip  = (__bf16*)carve((size_t)LAYERS * FD * FD * 2);
  __bf16* wrel   = (__bf16*)carve((size_t)LAYERS * FD * 2 * FD * 2);
  __bf16* wroot  = (__bf16*)carve((size_t)LAYERS * 2 * FD * FD * 2);

  auto blocks = [](long n, int t) { return (int)((n + t - 1) / t); };
  const dim3 gemm_blk(128);                              // 4 waves
  const int  gemm_gx = (NODES / 16 + 3) / 4;             // 3125 row tiles / 4 waves
  const int  nodeblk = (NODES + 3) / 4;                  // 1 wave per node kernels

  // ---- A-side bf16 conversion + B-side fragment packing ----
  k_cvt_bf16<<<blocks(ND, 256), 256, 0, stream>>>(x, x_bf, (int)ND);
  k_pack_b<<<blocks(FD * FD, 256), 256, 0, stream>>>(proj_W, wproj, FD, FD);
  for (int l = 0; l < LAYERS; ++l) {
    const size_t wo  = (size_t)l * FD * FD;
    const size_t wo2 = (size_t)l * FD * 2 * FD;
    k_pack_b<<<blocks(FD * FD, 256), 256, 0, stream>>>(gcn_W + wo, wgcn + wo, FD, FD);
    k_pack_b<<<blocks(FD * FD, 256), 256, 0, stream>>>(q_W + wo, wq + wo, FD, FD);
    k_pack_b<<<blocks(FD * FD, 256), 256, 0, stream>>>(k_W + wo, wk + wo, FD, FD);
    k_pack_b<<<blocks(FD * FD, 256), 256, 0, stream>>>(v_W + wo, wv + wo, FD, FD);
    k_pack_b<<<blocks(FD * FD, 256), 256, 0, stream>>>(skip_W + wo, wskip + wo, FD, FD);
    k_pack_b<<<blocks(FD * 2 * FD, 256), 256, 0, stream>>>(rel_W + wo2, wrel + wo2, FD, 2 * FD);
    k_pack_b<<<blocks(2 * FD * FD, 256), 256, 0, stream>>>(root_W + wo2, wroot + wo2, 2 * FD, FD);
  }

  // ---- input projection + BN + relu ----
  k_wmma_gemm<<<dim3(gemm_gx, 1), gemm_blk, 0, stream>>>(x_bf, wproj, proj_b, h, (__bf16*)nullptr,
                                                         NODES, FD, FD, 0);
  k_zero<<<1, 256, 0, stream>>>(stats, 2 * FD);
  k_colstats<<<256, FD, 0, stream>>>(h, NODES, stats, stats + FD);
  k_bn_relu<<<blocks(ND, 256), 256, 0, stream>>>(h, stats, stats + FD, bn_in_s, bn_in_b, h, h_bf, NODES);

  // ---- GCN symmetric norm (once) ----
  k_zero<<<blocks(NODES, 256), 256, 0, stream>>>(deg, NODES);
  k_deg<<<blocks(EDGES, 256), 256, 0, stream>>>(dst, deg);
  k_dis<<<blocks(NODES, 256), 256, 0, stream>>>(deg, dis);

  for (int l = 0; l < LAYERS; ++l) {
    const size_t wo  = (size_t)l * FD * FD;
    const size_t wo2 = (size_t)l * FD * 2 * FD;

    // local: hw = h @ gcn_W ; scatter with d^-1/2 norm
    k_wmma_gemm<<<dim3(gemm_gx, 1), gemm_blk, 0, stream>>>(h_bf, wgcn + wo, (const float*)nullptr,
                                                           hw, (__bf16*)nullptr, NODES, FD, FD, 0);
    k_zero<<<blocks(ND, 256), 256, 0, stream>>>(local, (int)ND);
    k_gcn_scatter<<<blocks((long)EDGES * 32, 256), 256, 0, stream>>>(hw, src, dst, dis, local);

    // global: q/k/v/skip projections
    k_wmma_gemm<<<dim3(gemm_gx, 1), gemm_blk, 0, stream>>>(h_bf, wq + wo, q_b + (size_t)l * FD,
                                                           qb_, (__bf16*)nullptr, NODES, FD, FD, 0);
    k_wmma_gemm<<<dim3(gemm_gx, 1), gemm_blk, 0, stream>>>(h_bf, wk + wo, k_b + (size_t)l * FD,
                                                           kb_, (__bf16*)nullptr, NODES, FD, FD, 0);
    k_wmma_gemm<<<dim3(gemm_gx, 1), gemm_blk, 0, stream>>>(h_bf, wv + wo, v_b + (size_t)l * FD,
                                                           vb_, (__bf16*)nullptr, NODES, FD, FD, 0);
    k_wmma_gemm<<<dim3(gemm_gx, 1), gemm_blk, 0, stream>>>(h_bf, wskip + wo, skip_b + (size_t)l * FD,
                                                           xr, (__bf16*)nullptr, NODES, FD, FD, 0);

    // attention: score -> segmax -> exp/denom -> agg
    k_fill_u32<<<blocks((long)NODES * HEADS, 256), 256, 0, stream>>>(smax, 0x007FFFFFu,
                                                                     NODES * HEADS); // enc(-inf)
    k_zero<<<blocks((long)NODES * HEADS, 256), 256, 0, stream>>>(denom, NODES * HEADS);
    k_attn_score<<<blocks((long)EDGES * HEADS, 256), 256, 0, stream>>>(qb_, kb_, src, dst, ex, smax);
    k_attn_exp<<<blocks((long)EDGES * HEADS, 256), 256, 0, stream>>>(ex, smax, dst, denom);
    k_zero<<<blocks(ND, 256), 256, 0, stream>>>(agg, (int)ND);
    k_attn_agg<<<blocks((long)EDGES * 32, 256), 256, 0, stream>>>(ex, denom, vb_, src, dst, agg);

    // beta gate -> glob (in agg); combine + LN1 -> y (+bf16)
    k_beta_glob<<<nodeblk, 128, 0, stream>>>(xr, agg, beta_W + (size_t)l * 3 * FD);
    k_combine_ln1<<<nodeblk, 128, 0, stream>>>(local, agg, gcn_b + (size_t)l * FD, lw, gw, l,
                                               ln1_s + (size_t)l * FD, ln1_b + (size_t)l * FD,
                                               y, y_bf);

    // FFN: f1 = relu(y @ rel_W) [N,256] ; f2 = f1 @ root_W [N,128]
    k_wmma_gemm<<<dim3(gemm_gx, 2), gemm_blk, 0, stream>>>(y_bf, wrel + wo2, (const float*)nullptr,
                                                           f1, f1_bf, NODES, FD, 2 * FD, 1);
    k_wmma_gemm<<<dim3(gemm_gx, 1), gemm_blk, 0, stream>>>(f1_bf, wroot + wo2, (const float*)nullptr,
                                                           f2, (__bf16*)nullptr, NODES, 2 * FD, FD, 0);
    k_add_ln2<<<nodeblk, 128, 0, stream>>>(f2, y, ln2_s + (size_t)l * FD, ln2_b + (size_t)l * FD);

    // model-level BN + relu -> h (+bf16)
    k_zero<<<1, 256, 0, stream>>>(stats, 2 * FD);
    k_colstats<<<256, FD, 0, stream>>>(f2, NODES, stats, stats + FD);
    k_bn_relu<<<blocks(ND, 256), 256, 0, stream>>>(f2, stats, stats + FD, bn_s + (size_t)l * FD,
                                                   bn_b + (size_t)l * FD, h, h_bf, NODES);
  }

  // ---- output head + log_softmax ----
  k_out<<<nodeblk, 128, 0, stream>>>(h, out_W, out_b, out);
}
